// DecoderModel_66829691126521
// MI455X (gfx1250) — matrix operand order repaired and checked by reading, not compiled, not run on hardware
//
#include <hip/hip_runtime.h>
#include <hip/hip_bf16.h>

// MI455X / gfx1250, wave32. D = A(16x32 bf16) x B(32x16 bf16) + C(16x16 f32)
typedef __attribute__((ext_vector_type(16))) __bf16 v16bf;
typedef __attribute__((ext_vector_type(8)))  float  v8f;

#define B_ 256
#define T_ 512
#define V_ 128
#define E_ 128
#define H_ 256

__device__ __forceinline__ unsigned short f2bf(float x) {
  union { float f; unsigned int u; } v; v.f = x;
  unsigned int u = v.u;
  u += 0x7FFFu + ((u >> 16) & 1u);           // round-to-nearest-even
  return (unsigned short)(u >> 16);
}

__device__ __forceinline__ float fast_rcp(float x) {
#if __has_builtin(__builtin_amdgcn_rcpf)
  return __builtin_amdgcn_rcpf(x);           // v_rcp_f32 (approx, ~1 ulp)
#else
  return 1.0f / x;
#endif
}

__device__ __forceinline__ float sigm(float x) {
  return fast_rcp(1.0f + __expf(-x));        // v_exp_f32 + v_add + v_rcp
}
__device__ __forceinline__ float tanh_fast(float x) {
  return 2.0f * fast_rcp(1.0f + __expf(-2.0f * x)) - 1.0f;
}

__device__ __forceinline__ v8f wmma_bf16(v16bf a, v16bf b, v8f c) {
  // (neg_a, A, neg_b, B, c_mod, C, reuse_a, reuse_b)
  return __builtin_amdgcn_wmma_f32_16x16x32_bf16(false, a, false, b, (short)0, c, false, false);
}

// ---------------------------------------------------------------------------
// Pack fp32 K x N weights into bf16 WMMA B-fragment tiles:
//   tile (kt,nt) of 32x16; lane = ((kk>>3)&1)*16 + nn; slot = (kk & 7) | ((kk & 16) >> 1)
// ---------------------------------------------------------------------------
__global__ void pack_weights_kernel(const float* __restrict__ W0,
                                    const float* __restrict__ W1,
                                    const float* __restrict__ Wd,
                                    unsigned short* __restrict__ w0pk,
                                    unsigned short* __restrict__ w1pk,
                                    unsigned short* __restrict__ wdpk) {
  const int t0 = 640 * 1024, t1 = 512 * 1024, td = 256 * 128;
  const int total = t0 + t1 + td;
  for (int idx = blockIdx.x * blockDim.x + threadIdx.x; idx < total;
       idx += gridDim.x * blockDim.x) {
    const float* src; unsigned short* dst; int k, n, NT, N;
    if (idx < t0)           { src = W0; dst = w0pk; k = idx / 1024;        n = idx % 1024;        NT = 64; N = 1024; }
    else if (idx < t0 + t1) { int i = idx - t0;      src = W1; dst = w1pk; k = i / 1024; n = i % 1024; NT = 64; N = 1024; }
    else                    { int i = idx - t0 - t1; src = Wd; dst = wdpk; k = i / 128;  n = i % 128;  NT = 8;  N = 128;  }
    int kt = k >> 5, kk = k & 31, nt = n >> 4, nn = n & 15;
    int lane = (((kk >> 3) & 1) << 4) + nn;
    int slot = (kk & 7) | ((kk & 16) >> 1);
    dst[(((kt * NT + nt) * 32) + lane) * 16 + slot] = f2bf(src[k * N + n]);
  }
}

// ---------------------------------------------------------------------------
// Persistent 2-layer LSTM + projection. One workgroup per 16 batch rows.
// 512 threads = 16 waves; wave w owns hidden units [w*16, w*16+16)
//   => gate N-tile for gate g is (g*16 + w); acc[] is only 4 fragments/wave.
// LDS aFrag tiles (A-fragment layout, 32 lanes x 16 bf16 each):
//   0..3 x_char(K0..127) | 4..11 cluster(K128..383) | 12..19 h0 | 20..27 h1
// ---------------------------------------------------------------------------
__global__ __launch_bounds__(512, 1)
void lstm_decoder_kernel(const int* __restrict__ tok,
                         const int* __restrict__ lens,
                         const int* __restrict__ clusIdx,
                         const float* __restrict__ charEmb,
                         const float* __restrict__ clusEmb,
                         const float* __restrict__ b0g,
                         const float* __restrict__ b1g,
                         const float* __restrict__ bDg,
                         const unsigned short* __restrict__ w0pk,
                         const unsigned short* __restrict__ w1pk,
                         const unsigned short* __restrict__ wdpk,
                         float* __restrict__ out)
{
  __shared__ __align__(32) unsigned short aFrag[28][32][16];
  __shared__ __align__(32) unsigned short outFrag[8][32][16];

  const int tid  = threadIdx.x;
  const int w    = tid >> 5;        // wave id 0..15
  const int lane = tid & 31;
  const int nn   = lane & 15;       // column within a 16-wide N tile
  const int b0r  = blockIdx.x * 16; // batch tile base row

  // zero h0/h1 state tiles (tiles 12..27)
  for (int i = tid; i < 16 * 32 * 16; i += 512)
    ((unsigned short*)aFrag[12])[i] = 0;

  // cluster embeddings (constant over t) -> tiles 4..11, A-fragment layout
  {
    int m  = tid >> 5;              // 0..15
    int k0 = lane * 8;              // 0..248
    int cl = clusIdx[b0r + m];
    const float* src = clusEmb + (size_t)cl * H_ + k0;
#pragma unroll
    for (int q = 0; q < 8; ++q) {
      int k = k0 + q, kk = k & 31;
      int dl   = (((kk >> 3) & 1) << 4) + m;
      int slot = (kk & 7) | ((kk & 16) >> 1);
      aFrag[4 + (k >> 5)][dl][slot] = f2bf(src[q]);
    }
  }

  // per-lane constants: biases for this lane's gate columns, row lengths
  float bias0[4], bias1[4];
#pragma unroll
  for (int g = 0; g < 4; ++g) {
    int n = g * H_ + w * 16 + nn;
    bias0[g] = b0g[n];
    bias1[g] = b1g[n];
  }
  float biasD = (w < 8) ? bDg[w * 16 + nn] : 0.f;

  int lenv[8];
#pragma unroll
  for (int r = 0; r < 8; ++r)
    lenv[r] = lens[b0r + ((lane < 16) ? r : 8 + r)];

  float c0v[8], c1v[8];
  unsigned short h0bf[8], h1bf[8];   // lane's stored bf16 h-state (for branchless blend)
#pragma unroll
  for (int r = 0; r < 8; ++r) { c0v[r] = 0.f; c1v[r] = 0.f; h0bf[r] = 0; h1bf[r] = 0; }

  // this lane's hidden-state element position (for h writes)
  const int kkh   = ((w & 1) << 4) | nn;                 // hcol % 32
  const int htile = w >> 1;                              // hcol / 32
  const int slotH = (kkh & 7) | ((kkh & 16) >> 1);
  const int dlHb  = ((kkh >> 3) & 1) << 4;               // dl = dlHb + m

  const v8f zero8 = {0.f, 0.f, 0.f, 0.f, 0.f, 0.f, 0.f, 0.f};

  __syncthreads();

  for (int t = 0; t < T_; ++t) {
    // Laundered zero offset in an SGPR: makes weight addresses formally
    // loop-variant (blocks LICM/spilling of B-fragments across the t-loop)
    // while the base pointers remain provably GLOBAL -> global_load_b128.
    int zoff = 0;
    asm volatile("" : "+s"(zoff));

    // ---- gather char embeddings for step t -> tiles 0..3 ----
    {
      int m  = tid >> 5;            // 0..15
      int k0 = lane * 4;            // 0..124
      int tk = tok[(b0r + m) * T_ + t];
      const float* src = charEmb + (size_t)tk * E_ + k0;
#pragma unroll
      for (int q = 0; q < 4; ++q) {
        int k = k0 + q, kk = k & 31;
        int dl   = (((kk >> 3) & 1) << 4) + m;
        int slot = (kk & 7) | ((kk & 16) >> 1);
        aFrag[k >> 5][dl][slot] = f2bf(src[q]);
      }
    }
    __syncthreads();

    // ---- layer 0 GEMM: z0 = [x|clus|h0](16x640) @ W0 ----
    v8f acc[4];
#pragma unroll
    for (int g = 0; g < 4; ++g) acc[g] = zero8;

#pragma clang loop unroll(disable)
    for (int kt = 0; kt < 20; ++kt) {
      v16bf a = *(const v16bf*)(&aFrag[kt][lane][0]);
#pragma unroll
      for (int g = 0; g < 4; ++g) {
        int nt = g * 16 + w;
        const v16bf b = *(const v16bf*)(w0pk + zoff + (((kt * 64 + nt) * 32) + lane) * 16);
        acc[g] = wmma_bf16(a, b, acc[g]);
      }
    }
    __syncthreads();  // all waves done reading old h0

    // ---- layer 0 LSTM cell (branchless mask); new h0 -> tiles 12..19 ----
#pragma unroll
    for (int r = 0; r < 8; ++r) {
      float iv = acc[0][r] + bias0[0];
      float jv = acc[1][r] + bias0[1];
      float fv = acc[2][r] + bias0[2];
      float ov = acc[3][r] + bias0[3];
      float cn = c0v[r] * sigm(fv + 1.0f) + sigm(iv) * tanh_fast(jv);
      float hn = tanh_fast(cn) * sigm(ov);
      bool msk = (t < lenv[r]);                 // copy-through when masked
      c0v[r] = msk ? cn : c0v[r];
      unsigned short hb = msk ? f2bf(hn) : h0bf[r];
      h0bf[r] = hb;
      int m = (lane < 16) ? r : 8 + r;
      aFrag[12 + htile][dlHb + m][slotH] = hb;  // unconditional store
    }
    __syncthreads();

    // ---- layer 1 GEMM: z1 = [h0|h1](16x512) @ W1 ----
#pragma unroll
    for (int g = 0; g < 4; ++g) acc[g] = zero8;

#pragma clang loop unroll(disable)
    for (int kt = 0; kt < 16; ++kt) {
      v16bf a = *(const v16bf*)(&aFrag[12 + kt][lane][0]);
#pragma unroll
      for (int g = 0; g < 4; ++g) {
        int nt = g * 16 + w;
        const v16bf b = *(const v16bf*)(w1pk + zoff + (((kt * 64 + nt) * 32) + lane) * 16);
        acc[g] = wmma_bf16(a, b, acc[g]);
      }
    }
    __syncthreads();  // all waves done reading old h1

    // ---- layer 1 LSTM cell; h1 state -> tiles 20..27, masked out -> outFrag ----
#pragma unroll
    for (int r = 0; r < 8; ++r) {
      float iv = acc[0][r] + bias1[0];
      float jv = acc[1][r] + bias1[1];
      float fv = acc[2][r] + bias1[2];
      float ov = acc[3][r] + bias1[3];
      float cn = c1v[r] * sigm(fv + 1.0f) + sigm(iv) * tanh_fast(jv);
      float hn = tanh_fast(cn) * sigm(ov);
      bool msk = (t < lenv[r]);
      c1v[r] = msk ? cn : c1v[r];
      unsigned short hb = msk ? f2bf(hn) : h1bf[r];
      h1bf[r] = hb;
      int m = (lane < 16) ? r : 8 + r;
      aFrag[20 + htile][dlHb + m][slotH] = hb;
      outFrag[htile][dlHb + m][slotH] = msk ? f2bf(hn) : (unsigned short)0;
    }
    __syncthreads();

    // ---- projection (waves 0..7): raw[b,t,:] = out_t(16x256) @ decoderW + bias ----
    if (w < 8) {
      v8f pacc = zero8;
#pragma clang loop unroll(disable)
      for (int kt = 0; kt < 8; ++kt) {
        v16bf a = *(const v16bf*)(&outFrag[kt][lane][0]);
        const v16bf b = *(const v16bf*)(wdpk + zoff + (((kt * 8 + w) * 32) + lane) * 16);
        pacc = wmma_bf16(a, b, pacc);
      }
      int n = w * 16 + nn;
#pragma unroll
      for (int r = 0; r < 8; ++r) {
        int m = (lane < 16) ? r : 8 + r;
        out[((size_t)(b0r + m) * T_ + t) * V_ + n] = pacc[r] + biasD;
      }
    }
    __syncthreads();  // outFrag rewritten next iteration
  }
}

// ---------------------------------------------------------------------------
extern "C" void kernel_launch(void* const* d_in, const int* in_sizes, int n_in,
                              void* d_out, int out_size, void* d_ws, size_t ws_size,
                              hipStream_t stream) {
  const int*   tok     = (const int*)d_in[0];   // [B,T]
  const int*   lens    = (const int*)d_in[1];   // [B]
  const int*   clus    = (const int*)d_in[2];   // [B]
  const float* charEmb = (const float*)d_in[3]; // [V,E]
  const float* clusEmb = (const float*)d_in[4]; // [NC,H]
  const float* W0      = (const float*)d_in[5]; // [640,1024]
  const float* b0      = (const float*)d_in[6]; // [1024]
  const float* W1      = (const float*)d_in[7]; // [512,1024]
  const float* b1      = (const float*)d_in[8]; // [1024]
  const float* Wd      = (const float*)d_in[9]; // [256,128]
  const float* bD      = (const float*)d_in[10];// [128]

  unsigned short* w0pk = (unsigned short*)d_ws;       // 640*1024 bf16
  unsigned short* w1pk = w0pk + 640 * 1024;           // 512*1024 bf16
  unsigned short* wdpk = w1pk + 512 * 1024;           // 256*128 bf16 (~2.4 MB total)

  pack_weights_kernel<<<512, 256, 0, stream>>>(W0, W1, Wd, w0pk, w1pk, wdpk);
  lstm_decoder_kernel<<<B_ / 16, 512, 0, stream>>>(tok, lens, clus, charEmb, clusEmb,
                                                   b0, b1, bD, w0pk, w1pk, wdpk,
                                                   (float*)d_out);
}
